// MusicTransformer_6167573037617
// MI455X (gfx1250) — matrix operand order repaired and checked by reading, compiled, not run on hardware
//
#include <hip/hip_runtime.h>

// ---------------------------------------------------------------------------
// Music Transformer forward for MI455X (gfx1250, wave32, WMMA f16->f32).
// ---------------------------------------------------------------------------

typedef __attribute__((ext_vector_type(16))) _Float16 v16h;
typedef __attribute__((ext_vector_type(8)))  _Float16 v8h;
typedef __attribute__((ext_vector_type(4)))  _Float16 v4h;
typedef __attribute__((ext_vector_type(8)))  float    v8f;

#define NLAYER 6
#define BB 2
#define LL 1024
#define DD 256
#define HH 4
#define DHD 64
#define VV 240
#define BL (BB * LL)

#define LDSTR 40   // LDS row stride in halfs: multiple of 8 (16B align), skews banks

// ---------------------------------------------------------------------------
// Generic batched GEMM:  C = act(A @ B + bias) (+ residual)
//   A: (M,K) f32, element (m,k) = A[aoff + m*lda + k]
//   B: BT==0 -> element(k,n) = Bm[boff + k*ldb + n]
//      BT==1 -> element(k,n) = Bm[boff + n*ldb + k]
// Batch z decomposes as (z/nH, z%nH) with separate strides (covers b,h).
// Block = 128 threads = 4 waves; block tile 64(M) x 64(N); each wave owns a
// 32x32 C tile -> 4 v_wmma_f32_16x16x32_f16 per K-step with A/B frag reuse.
// Software-pipelined: tile t+1 is fetched to registers while WMMA runs on
// tile t (double-buffered LDS, one barrier per K-step); the loadcnt wait
// lands after the WMMAs so HBM/L2 latency hides under matrix math.
// Requires M%64==0, N%64==0, K%32==0 (true for every call in this model).
// ---------------------------------------------------------------------------
template<int BT>
__global__ __launch_bounds__(128) void gemm_wmma_f16(
    const float* __restrict__ A, long long saB, long long saH, int lda,
    const float* __restrict__ Bm, long long sbB, long long sbH, int ldb,
    const float* __restrict__ bias,
    const float* __restrict__ Rs, long long srB, long long srH, int ldr,
    float* __restrict__ C, long long scB, long long scH, int ldc,
    int nH, int M, int N, int K, int act)
{
    __shared__ _Float16 As[2][64 * LDSTR];   // A tile: [m][k], 64x32 used
    __shared__ _Float16 Bs[2][64 * LDSTR];   // B tile: [n][k], 64x32 used

    const int tid  = threadIdx.x;
    const int lane = tid & 31;
    const int wave = tid >> 5;
    const int wr   = wave >> 1;           // 0..1  (M direction)
    const int wc   = wave & 1;            // 0..1  (N direction)
    const int nn   = lane & 15;
    const int half = lane >> 4;

    const int z  = blockIdx.z;
    const int zb = z / nH;
    const int zh = z % nH;

    const long long aoff = (long long)zb * saB + (long long)zh * saH;
    const long long boff = (long long)zb * sbB + (long long)zh * sbH;
    const long long coff = (long long)zb * scB + (long long)zh * scH;

    const int m0 = blockIdx.y * 64;
    const int n0 = blockIdx.x * 64;

    // Per-thread staging coordinates (compile-time-regular).
    const int am  = tid >> 3;             // A row base (+16c), c=0..3
    const int ak4 = (tid & 7) * 4;        // A k offset (float4)
    const int bk  = tid >> 4;             // BT==0: B k-row base (+8c)
    const int bn4 = (tid & 15) * 4;       // BT==0: B n offset (float4)

    float4 ar[4], br[4];

    auto fetch = [&](int k0) {
        #pragma unroll
        for (int c = 0; c < 4; ++c)
            ar[c] = *(const float4*)(A + aoff
                      + (long long)(m0 + am + 16 * c) * lda + (k0 + ak4));
        if (BT) {
            #pragma unroll
            for (int c = 0; c < 4; ++c)
                br[c] = *(const float4*)(Bm + boff
                          + (long long)(n0 + am + 16 * c) * ldb + (k0 + ak4));
        } else {
            #pragma unroll
            for (int c = 0; c < 4; ++c)
                br[c] = *(const float4*)(Bm + boff
                          + (long long)(k0 + bk + 8 * c) * ldb + (n0 + bn4));
        }
    };
    auto commit = [&](int buf) {
        #pragma unroll
        for (int c = 0; c < 4; ++c) {
            v4h h;
            h[0] = (_Float16)ar[c].x; h[1] = (_Float16)ar[c].y;
            h[2] = (_Float16)ar[c].z; h[3] = (_Float16)ar[c].w;
            *(v4h*)&As[buf][(am + 16 * c) * LDSTR + ak4] = h;
        }
        if (BT) {
            #pragma unroll
            for (int c = 0; c < 4; ++c) {
                v4h h;
                h[0] = (_Float16)br[c].x; h[1] = (_Float16)br[c].y;
                h[2] = (_Float16)br[c].z; h[3] = (_Float16)br[c].w;
                *(v4h*)&Bs[buf][(am + 16 * c) * LDSTR + ak4] = h;
            }
        } else {
            #pragma unroll
            for (int c = 0; c < 4; ++c) {
                int k = bk + 8 * c;
                Bs[buf][(bn4 + 0) * LDSTR + k] = (_Float16)br[c].x;
                Bs[buf][(bn4 + 1) * LDSTR + k] = (_Float16)br[c].y;
                Bs[buf][(bn4 + 2) * LDSTR + k] = (_Float16)br[c].z;
                Bs[buf][(bn4 + 3) * LDSTR + k] = (_Float16)br[c].w;
            }
        }
    };

    v8f acc[2][2];
    #pragma unroll
    for (int i = 0; i < 2; ++i)
        #pragma unroll
        for (int j = 0; j < 2; ++j)
            acc[i][j] = (v8f){0.f, 0.f, 0.f, 0.f, 0.f, 0.f, 0.f, 0.f};

    const int T = K / 32;
    fetch(0);
    commit(0);

    for (int t = 0; t < T; ++t) {
        const int buf = t & 1;
        const bool more = (t + 1 < T);
        if (more) {
            fetch((t + 1) * 32);          // issue loads; no wait yet
            __builtin_prefetch(A + aoff + (long long)(m0 + am) * lda
                               + (t + 2) * 32 + ak4, 0, 0);
        }
        __syncthreads();                  // LDS tile `buf` ready

        // ---- fragments (contiguous LDS runs per ISA 7.12.2 layouts) ----
        v16h af[2], bf[2];
        #pragma unroll
        for (int i = 0; i < 2; ++i) {
            const _Float16* ap = &As[buf][(wr * 32 + i * 16 + nn) * LDSTR];
            union { v16h v; v8h h[2]; } u;
            u.h[0] = *(const v8h*)(ap + half * 8);        // K = half*8 .. +7
            u.h[1] = *(const v8h*)(ap + 16 + half * 8);   // K = 16+half*8 .. +7
            af[i] = u.v;
        }
        #pragma unroll
        for (int j = 0; j < 2; ++j) {
            const _Float16* bp = &Bs[buf][(wc * 32 + j * 16 + nn) * LDSTR];
            union { v16h v; v8h h[2]; } u;
            u.h[0] = *(const v8h*)(bp + half * 8);        // K = half*8 .. +7
            u.h[1] = *(const v8h*)(bp + 16 + half * 8);   // K = 16+half*8 .. +7
            bf[j] = u.v;
        }
        #pragma unroll
        for (int i = 0; i < 2; ++i)
            #pragma unroll
            for (int j = 0; j < 2; ++j)
                acc[i][j] = __builtin_amdgcn_wmma_f32_16x16x32_f16(
                    false, af[i], false, bf[j], (short)0, acc[i][j],
                    false, false);

        if (more) commit((t + 1) & 1);    // loadcnt wait lands after WMMAs
    }

    // ---- epilogue: bias / activation / residual, 16-lane coalesced rows ----
    #pragma unroll
    for (int i = 0; i < 2; ++i) {
        #pragma unroll
        for (int j = 0; j < 2; ++j) {
            #pragma unroll
            for (int r = 0; r < 8; ++r) {
                int mrow = m0 + wr * 32 + i * 16 + half * 8 + r;
                int ncol = n0 + wc * 32 + j * 16 + nn;
                float v = acc[i][j][r];
                if (bias) v += bias[ncol];
                if (act == 1) v = (v > 0.f) ? v : 0.2f * v;   // leaky relu 0.2
                if (Rs) v += Rs[(long long)zb * srB + (long long)zh * srH
                                + (long long)mrow * ldr + ncol];
                C[coff + (long long)mrow * ldc + ncol] = v;
            }
        }
    }
}

// ---------------------------------------------------------------------------
// x[b,l,:] = emb_table[tokens[b,l], :] + pos_emb[l, :]
// ---------------------------------------------------------------------------
__global__ void embed_kernel(const int* __restrict__ tokens,
                             const float* __restrict__ emb,
                             const float* __restrict__ pos,
                             float* __restrict__ x)
{
    int i = blockIdx.x * blockDim.x + threadIdx.x;
    if (i >= BB * LL * DD) return;
    int d  = i % DD;
    int bl = i / DD;
    int l  = bl % LL;
    int tok = tokens[bl];
    x[i] = emb[tok * DD + d] + pos[l * DD + d];
}

// ---------------------------------------------------------------------------
// logits[b,h,l,m] = (QKt[b,h,l,m] + Srel[b,h,l,m]) * (1/sqrt(dh))
// Srel via the exact pad/reshape skew of the reference, with the
// (idx_m >= L-1-idx_l) mask applied on the QE read.
// ---------------------------------------------------------------------------
__global__ void skew_kernel(const float* __restrict__ QKt,
                            const float* __restrict__ QE,
                            float* __restrict__ out)
{
    long long i = (long long)blockIdx.x * 256 + threadIdx.x;
    if (i >= (long long)BB * HH * LL * LL) return;
    int m       = (int)(i & (LL - 1));
    long long r = i >> 10;
    int l       = (int)(r & (LL - 1));
    long long bh = r >> 10;

    int t  = (l + 1) * LL + m;
    int lq = t / (LL + 1);
    int j  = t % (LL + 1);
    float srel = 0.f;
    if (j >= 1) {
        int mq = j - 1;
        if (mq >= LL - 1 - lq)
            srel = QE[(bh << 20) + ((long long)lq << 10) + mq];
    }
    out[i] = (QKt[i] + srel) * 0.125f;   // 1/sqrt(64)
}

// ---------------------------------------------------------------------------
// In-place softmax over rows of length LL (one block / row).
// ---------------------------------------------------------------------------
__global__ void softmax_rows(float* __restrict__ data)
{
    __shared__ float red[256];
    float* p = data + (long long)blockIdx.x * LL;
    int t = threadIdx.x;

    float mx = -3.4e38f;
    for (int i = t; i < LL; i += 256) mx = fmaxf(mx, p[i]);
    red[t] = mx; __syncthreads();
    for (int s = 128; s > 0; s >>= 1) {
        if (t < s) red[t] = fmaxf(red[t], red[t + s]);
        __syncthreads();
    }
    mx = red[0]; __syncthreads();

    float sum = 0.f;
    for (int i = t; i < LL; i += 256) {
        float e = __expf(p[i] - mx);
        p[i] = e;
        sum += e;
    }
    red[t] = sum; __syncthreads();
    for (int s = 128; s > 0; s >>= 1) {
        if (t < s) red[t] += red[t + s];
        __syncthreads();
    }
    float inv = 1.f / red[0];
    for (int i = t; i < LL; i += 256) p[i] *= inv;
}

// ---------------------------------------------------------------------------
// Classifier: split-K over the 252 MB fc_W stream (coalesced across N),
// atomic partial reduction into out_pre[2*240].
// ---------------------------------------------------------------------------
__global__ void zero_kernel(float* __restrict__ p, int n)
{
    int i = blockIdx.x * blockDim.x + threadIdx.x;
    if (i < n) p[i] = 0.f;
}

__global__ void fc_splitk(const float* __restrict__ x,   // (B, L*D) contiguous
                          const float* __restrict__ W,   // (L*D, V)
                          float* __restrict__ out_pre,   // (B, V)
                          int K, int KC)
{
    int n = threadIdx.x;
    if (n >= VV) return;
    int k0 = blockIdx.x * KC;
    float a0 = 0.f, a1 = 0.f;
    for (int i = 0; i < KC; ++i) {
        int k = k0 + i;
        float w = W[(long long)k * VV + n];
        a0 += x[k] * w;
        a1 += x[K + k] * w;
    }
    atomicAdd(&out_pre[n], a0);
    atomicAdd(&out_pre[VV + n], a1);
}

__global__ void softmax_out(const float* __restrict__ pre,
                            const float* __restrict__ bias,
                            float* __restrict__ out)
{
    __shared__ float red[256];
    int b = blockIdx.x;
    int t = threadIdx.x;
    float v = (t < VV) ? pre[b * VV + t] + bias[t] : -3.4e38f;
    red[t] = v; __syncthreads();
    for (int s = 128; s > 0; s >>= 1) {
        if (t < s) red[t] = fmaxf(red[t], red[t + s]);
        __syncthreads();
    }
    float mx = red[0]; __syncthreads();
    float e = (t < VV) ? __expf(v - mx) : 0.f;
    red[t] = e; __syncthreads();
    for (int s = 128; s > 0; s >>= 1) {
        if (t < s) red[t] += red[t + s];
        __syncthreads();
    }
    float inv = 1.f / red[0];
    if (t < VV) out[b * VV + t] = e * inv;
}

// ---------------------------------------------------------------------------
// Host-side orchestration.
// ---------------------------------------------------------------------------
extern "C" void kernel_launch(void* const* d_in, const int* in_sizes, int n_in,
                              void* d_out, int out_size, void* d_ws, size_t ws_size,
                              hipStream_t stream)
{
    (void)in_sizes; (void)n_in; (void)out_size; (void)ws_size;

    const int*   tokens = (const int*)  d_in[0];
    const float* emb    = (const float*)d_in[1];
    const float* pos    = (const float*)d_in[2];
    const float* Wq     = (const float*)d_in[3];
    const float* bq     = (const float*)d_in[4];
    const float* Wk     = (const float*)d_in[5];
    const float* bk     = (const float*)d_in[6];
    const float* Wv     = (const float*)d_in[7];
    const float* bv     = (const float*)d_in[8];
    const float* Wo     = (const float*)d_in[9];
    const float* bo     = (const float*)d_in[10];
    const float* E      = (const float*)d_in[11];
    const float* W1     = (const float*)d_in[12];
    const float* b1     = (const float*)d_in[13];
    const float* W2     = (const float*)d_in[14];
    const float* b2     = (const float*)d_in[15];
    const float* fcW    = (const float*)d_in[16];
    const float* fcb    = (const float*)d_in[17];

    // Workspace layout (floats). Total ~20.45M floats (~82 MB).
    float* ws = (float*)d_ws;
    const size_t XSZ = (size_t)BB * LL * DD;        // 524288
    const size_t SSZ = (size_t)BB * HH * LL * LL;   // 8388608
    float* qb  = ws;
    float* kb  = qb  + XSZ;
    float* vb  = kb  + XSZ;
    float* qe  = vb  + XSZ;
    float* sc  = qe  + SSZ;
    float* ao  = sc  + SSZ;
    float* xA  = ao  + XSZ;
    float* xB  = xA  + XSZ;
    float* hb  = xB  + XSZ;
    float* pre = hb  + XSZ;

    const long long LD  = (long long)LL * DD;       // per-batch stride of (B,L,D)
    const long long LLl = (long long)LL * LL;       // per-head stride of scores
    const long long HLL = (long long)HH * LLl;      // per-batch stride of scores

    auto gemm = [&](const float* A, long long saB, long long saH, int lda,
                    const float* Bm, long long sbB, long long sbH, int ldb, int bT,
                    const float* bias, const float* Rs, long long srB, long long srH,
                    int ldr, float* C, long long scB, long long scH, int ldc,
                    int nH, int M, int N, int K, int act, int batches) {
        dim3 grid(N / 64, M / 64, batches);
        if (bT)
            gemm_wmma_f16<1><<<grid, 128, 0, stream>>>(
                A, saB, saH, lda, Bm, sbB, sbH, ldb, bias,
                Rs, srB, srH, ldr, C, scB, scH, ldc, nH, M, N, K, act);
        else
            gemm_wmma_f16<0><<<grid, 128, 0, stream>>>(
                A, saB, saH, lda, Bm, sbB, sbH, ldb, bias,
                Rs, srB, srH, ldr, C, scB, scH, ldc, nH, M, N, K, act);
    };

    // Embedding + positional encoding.
    embed_kernel<<<(BB * LL * DD + 255) / 256, 256, 0, stream>>>(tokens, emb, pos, xA);

    float* cur = xA;
    float* alt = xB;

    for (int l = 0; l < NLAYER; ++l) {
        for (int s = 0; s < 2; ++s) {
            const int ls = l * 2 + s;
            const float* wq = Wq + (size_t)ls * DD * DD;
            const float* wk = Wk + (size_t)ls * DD * DD;
            const float* wv = Wv + (size_t)ls * DD * DD;
            const float* wo = Wo + (size_t)ls * DD * DD;
            const float* el = E  + (size_t)ls * LL * DHD;

            // Q/K/V projections: (2048,256) @ (256,256) + bias
            gemm(cur, 0, 0, DD, wq, 0, 0, DD, 0, bq + ls * DD,
                 nullptr, 0, 0, 0, qb, 0, 0, DD, 1, BL, DD, DD, 0, 1);
            gemm(cur, 0, 0, DD, wk, 0, 0, DD, 0, bk + ls * DD,
                 nullptr, 0, 0, 0, kb, 0, 0, DD, 1, BL, DD, DD, 0, 1);
            gemm(cur, 0, 0, DD, wv, 0, 0, DD, 0, bv + ls * DD,
                 nullptr, 0, 0, 0, vb, 0, 0, DD, 1, BL, DD, DD, 0, 1);

            // QE[b,h,l,m] = q . Erel[m]   (B=E, transposed, shared over batch)
            gemm(qb, LD, DHD, DD, el, 0, 0, DHD, 1, nullptr,
                 nullptr, 0, 0, 0, qe, HLL, LLl, LL, HH, LL, LL, DHD, 0, BB * HH);

            // QKt[b,h,l,m] = q . k[m]     (B=k, transposed)
            gemm(qb, LD, DHD, DD, kb, LD, DHD, DD, 1, nullptr,
                 nullptr, 0, 0, 0, sc, HLL, LLl, LL, HH, LL, LL, DHD, 0, BB * HH);

            // logits = (QKt + skew(mask(QE))) / sqrt(dh), in place on sc.
            skew_kernel<<<(unsigned)(SSZ / 256), 256, 0, stream>>>(sc, qe, sc);

            // softmax over last dim.
            softmax_rows<<<BB * HH * LL, 256, 0, stream>>>(sc);

            // out[b,h,l,d] = attn @ v
            gemm(sc, HLL, LLl, LL, vb, LD, DHD, DD, 0, nullptr,
                 nullptr, 0, 0, 0, ao, LD, DHD, DD, HH, LL, DHD, LL, 0, BB * HH);

            // output projection + bias + residual.
            gemm(ao, 0, 0, DD, wo, 0, 0, DD, 0, bo + ls * DD,
                 cur, 0, 0, DD, alt, 0, 0, DD, 1, BL, DD, DD, 0, 1);

            float* t = cur; cur = alt; alt = t;
        }

        // FFN: h = leaky_relu(x@W1+b1, 0.2); x = h@W2+b2
        gemm(cur, 0, 0, DD, W1 + (size_t)l * DD * DD, 0, 0, DD, 0, b1 + l * DD,
             nullptr, 0, 0, 0, hb, 0, 0, DD, 1, BL, DD, DD, 1, 1);
        gemm(hb, 0, 0, DD, W2 + (size_t)l * DD * DD, 0, 0, DD, 0, b2 + l * DD,
             nullptr, 0, 0, 0, alt, 0, 0, DD, 1, BL, DD, DD, 0, 1);
        float* t = cur; cur = alt; alt = t;
    }

    // Classifier: flat(B, L*D) @ fc_W(L*D, V) + fc_b, softmax.
    zero_kernel<<<2, 256, 0, stream>>>(pre, 2 * VV);
    fc_splitk<<<128, 256, 0, stream>>>(cur, fcW, pre, LL * DD, (LL * DD) / 128);
    softmax_out<<<2, 256, 0, stream>>>(pre, fcb, (float*)d_out);
}
